// SlowMultiHeadAttention_59201829208706
// MI455X (gfx1250) — matrix operand order repaired and checked by reading, not compiled, not run on hardware
//
#include <hip/hip_runtime.h>
#include <hip/hip_bf16.h>

#define S_LEN 2048
#define BATCH 4
#define DMODEL 1024
#define NH 16
#define DHEAD 64
#define ATT_SCALE 0.125f  // 1/sqrt(64)

typedef __attribute__((ext_vector_type(16))) _Float16 v16h;
typedef __attribute__((ext_vector_type(8)))  _Float16 v8h;
typedef __attribute__((ext_vector_type(8)))  float    v8f;

// ---------------------------------------------------------------------------
// Fragment loaders for V_WMMA_F32_16X16X32_F16 (wave32).
// A (16x32, MxK): lane = r + 16*h holds row r; elements j<8 -> K = 8h + j,
//                 elements j>=8 -> K = 16 + 8h + (j-8).  (ISA 7.12.2)
// B (32x16, KxN): lane = n + 16*h holds col n; same K mapping as A.
// Both are contiguous 16B loads from row-major storage.
// ---------------------------------------------------------------------------
__device__ inline v16h load_a_frag(const _Float16* base, int ld, int row0, int col0, int lane) {
    int r = lane & 15, h = lane >> 4;
    const _Float16* p = base + (size_t)(row0 + r) * ld + col0 + 8 * h;
    v8h lo = *(const v8h*)(p);
    v8h hi = *(const v8h*)(p + 16);
    v16h a;
#pragma unroll
    for (int i = 0; i < 8; ++i) { a[i] = lo[i]; a[8 + i] = hi[i]; }
    return a;
}

// W is row-major [N][K]; computes fragment of B where B[k][n] = W[n][k] (i.e. X @ W^T)
__device__ inline v16h load_b_frag(const _Float16* w, int ld, int n0, int k0, int lane) {
    int n = lane & 15, h = lane >> 4;
    const _Float16* p = w + (size_t)(n0 + n) * ld + k0 + 8 * h;
    v8h lo = *(const v8h*)(p);
    v8h hi = *(const v8h*)(p + 16);
    v16h b;
#pragma unroll
    for (int i = 0; i < 8; ++i) { b[i] = lo[i]; b[8 + i] = hi[i]; }
    return b;
}

__device__ inline v8f wmma_f16(v16h a, v16h b, v8f c) {
    return __builtin_amdgcn_wmma_f32_16x16x32_f16(false, a, false, b, (short)0, c, false, false);
}

// ---------------------------------------------------------------------------
// Prep: f32 -> f16 conversions
// ---------------------------------------------------------------------------
__global__ void cvt_f16_kernel(const float* __restrict__ src, _Float16* __restrict__ dst, int n) {
    int i = blockIdx.x * blockDim.x + threadIdx.x;
    if (i < n) dst[i] = (_Float16)src[i];
}

// inp [S,B,D] f32 -> x16 [B,S,D] f16
__global__ void transpose_x_kernel(const float* __restrict__ inp, _Float16* __restrict__ x16) {
    int i = blockIdx.x * blockDim.x + threadIdx.x;
    if (i >= S_LEN * BATCH * DMODEL) return;
    int d = i % DMODEL;
    int rem = i / DMODEL;
    int b = rem % BATCH;
    int s = rem / BATCH;
    x16[((size_t)b * S_LEN + s) * DMODEL + d] = (_Float16)inp[i];
}

// ---------------------------------------------------------------------------
// QKV projection: C[s, j] = x16[b] @ Wqkv^T + bqkv  (M=S, N=3D, K=D)
// q,k stored [b,h,s,dh]; v stored transposed [b,h,dh,s]
// ---------------------------------------------------------------------------
__global__ __launch_bounds__(32) void qkv_gemm_kernel(
    const _Float16* __restrict__ x16, const _Float16* __restrict__ w16,
    const float* __restrict__ bias,
    _Float16* __restrict__ q16, _Float16* __restrict__ k16, _Float16* __restrict__ vT16) {
    int lane = threadIdx.x;
    int idx = blockIdx.x;
    int st = idx & 127;            // S/16
    int jt = (idx >> 7) % 192;     // 3D/16
    int b  = idx / (128 * 192);
    int s0 = st * 16, j0 = jt * 16;
    const _Float16* A = x16 + (size_t)b * S_LEN * DMODEL;

    v8f c = {};
    for (int kc = 0; kc < DMODEL; kc += 32) {
        v16h a = load_a_frag(A, DMODEL, s0, kc, lane);
        v16h w = load_b_frag(w16, DMODEL, j0, kc, lane);
        c = wmma_f16(a, w, c);
    }
    int n = lane & 15, hh = lane >> 4;
    float bj = bias[j0 + n];
#pragma unroll
    for (int r = 0; r < 8; ++r) c[r] += bj;

    int part = j0 >> 10;           // 0=q 1=k 2=v
    int jj = j0 & 1023;
    int head = jj >> 6;
    int dh0 = jj & 63;
    size_t bh = (size_t)b * NH + head;

    if (part == 2) {
        // vT16[bh][dh0+n][s0 + 8*hh + r]  : 8 f16 contiguous in s
        v8h pk;
#pragma unroll
        for (int r = 0; r < 8; ++r) pk[r] = (_Float16)c[r];
        size_t addr = (bh * DHEAD + dh0 + n) * S_LEN + s0 + 8 * hh;
        *(v8h*)(vT16 + addr) = pk;
    } else {
        _Float16* dst = (part == 0) ? q16 : k16;
#pragma unroll
        for (int r = 0; r < 8; ++r) {
            size_t s = (size_t)(s0 + r + 8 * hh);
            dst[(bh * S_LEN + s) * DHEAD + dh0 + n] = (_Float16)c[r];
        }
    }
}

// ---------------------------------------------------------------------------
// Flash attention: one wave per (b, h, 16 query rows); online softmax over
// 32-key tiles. C-fragment row M = vgpr + 8*(lane/16), col N = lane%16.
// ---------------------------------------------------------------------------
__global__ __launch_bounds__(32) void flash_attn_kernel(
    const _Float16* __restrict__ q16, const _Float16* __restrict__ k16,
    const _Float16* __restrict__ vT16, const int* __restrict__ attn_mask,
    _Float16* __restrict__ attn16) {
    __shared__ _Float16 lp[32 * 16];   // column-major tile: [32 cols][16 rows]

    int lane = threadIdx.x;
    int idx = blockIdx.x;
    int qt = idx & 127;
    int head = (idx >> 7) & 15;
    int b = idx >> 11;
    int s0 = qt * 16;
    size_t bh = (size_t)b * NH + head;
    const _Float16* Q = q16 + bh * S_LEN * DHEAD;
    const _Float16* K = k16 + bh * S_LEN * DHEAD;
    const _Float16* V = vT16 + bh * DHEAD * S_LEN;
    const int* msk = attn_mask + b * S_LEN;
    int n = lane & 15, hh = lane >> 4;

    v16h aq0 = load_a_frag(Q, DHEAD, s0, 0, lane);
    v16h aq1 = load_a_frag(Q, DHEAD, s0, 32, lane);

    v8f o0 = {}, o1 = {}, o2 = {}, o3 = {};
    float mrow[8], lrow[8];
#pragma unroll
    for (int r = 0; r < 8; ++r) { mrow[r] = -INFINITY; lrow[r] = 0.0f; }

    for (int k0 = 0; k0 < S_LEN; k0 += 32) {
        // ---- scores for 32 keys: two 16x16 C fragments ----
        v8f sa = {}, sb = {};
        sa = wmma_f16(aq0, load_b_frag(K, DHEAD, k0, 0, lane), sa);
        sa = wmma_f16(aq1, load_b_frag(K, DHEAD, k0, 32, lane), sa);
        sb = wmma_f16(aq0, load_b_frag(K, DHEAD, k0 + 16, 0, lane), sb);
        sb = wmma_f16(aq1, load_b_frag(K, DHEAD, k0 + 16, 32, lane), sb);

        // scale + key-padding mask (reference masks where attn_mask != 0)
        bool m0 = msk[k0 + n] != 0;
        bool m1 = msk[k0 + 16 + n] != 0;
#pragma unroll
        for (int r = 0; r < 8; ++r) {
            sa[r] = m0 ? -INFINITY : sa[r] * ATT_SCALE;
            sb[r] = m1 ? -INFINITY : sb[r] * ATT_SCALE;
        }

        // ---- row max over the 32 columns ----
        float rm[8];
#pragma unroll
        for (int r = 0; r < 8; ++r) rm[r] = fmaxf(sa[r], sb[r]);
#pragma unroll
        for (int d = 1; d < 16; d <<= 1) {
#pragma unroll
            for (int r = 0; r < 8; ++r) rm[r] = fmaxf(rm[r], __shfl_xor(rm[r], d, 32));
        }

        // ---- online softmax update ----
        float rs[8];
#pragma unroll
        for (int r = 0; r < 8; ++r) {
            float mn = fmaxf(mrow[r], rm[r]);
            float corr = (mrow[r] == -INFINITY) ? 0.0f : expf(mrow[r] - mn);
            float p0 = (mn == -INFINITY) ? 0.0f : expf(sa[r] - mn);
            float p1 = (mn == -INFINITY) ? 0.0f : expf(sb[r] - mn);
            sa[r] = p0; sb[r] = p1;
            rs[r] = p0 + p1;
            mrow[r] = mn;
            lrow[r] *= corr;
            o0[r] *= corr; o1[r] *= corr; o2[r] *= corr; o3[r] *= corr;
        }
#pragma unroll
        for (int d = 1; d < 16; d <<= 1) {
#pragma unroll
            for (int r = 0; r < 8; ++r) rs[r] += __shfl_xor(rs[r], d, 32);
        }
#pragma unroll
        for (int r = 0; r < 8; ++r) lrow[r] += rs[r];

        // ---- P (C layout) -> LDS column-major -> A fragment ----
        v8h p0h, p1h;
#pragma unroll
        for (int r = 0; r < 8; ++r) { p0h[r] = (_Float16)sa[r]; p1h[r] = (_Float16)sb[r]; }
        *(v8h*)&lp[n * 16 + 8 * hh] = p0h;            // cols 0..15
        *(v8h*)&lp[(n + 16) * 16 + 8 * hh] = p1h;     // cols 16..31
        __syncthreads();
        v16h aP;
#pragma unroll
        for (int j = 0; j < 8; ++j) {
            aP[j]     = lp[(8 * hh + j) * 16 + n];        // K = 8h + j
            aP[8 + j] = lp[(16 + 8 * hh + j) * 16 + n];   // K = 16 + 8h + j
        }
        __syncthreads();

        // ---- O += P @ V  (B fragments from V^T rows, contiguous) ----
        o0 = wmma_f16(aP, load_b_frag(V, S_LEN, 0,  k0, lane), o0);
        o1 = wmma_f16(aP, load_b_frag(V, S_LEN, 16, k0, lane), o1);
        o2 = wmma_f16(aP, load_b_frag(V, S_LEN, 32, k0, lane), o2);
        o3 = wmma_f16(aP, load_b_frag(V, S_LEN, 48, k0, lane), o3);
    }

    // ---- normalize and store attn16 [B,S,D], d = head*64 + 16c + n ----
#pragma unroll
    for (int r = 0; r < 8; ++r) {
        float inv = 1.0f / lrow[r];
        o0[r] *= inv; o1[r] *= inv; o2[r] *= inv; o3[r] *= inv;
    }
#pragma unroll
    for (int r = 0; r < 8; ++r) {
        size_t row = ((size_t)b * S_LEN + s0 + r + 8 * hh) * DMODEL + head * DHEAD + n;
        attn16[row +  0] = (_Float16)o0[r];
        attn16[row + 16] = (_Float16)o1[r];
        attn16[row + 32] = (_Float16)o2[r];
        attn16[row + 48] = (_Float16)o3[r];
    }
}

// ---------------------------------------------------------------------------
// Output projection: out[s,b,j] = attn16[b] @ Wproj^T + bproj   (f32 out)
// ---------------------------------------------------------------------------
__global__ __launch_bounds__(32) void proj_gemm_kernel(
    const _Float16* __restrict__ attn16, const _Float16* __restrict__ wp16,
    const float* __restrict__ bias, float* __restrict__ out) {
    int lane = threadIdx.x;
    int idx = blockIdx.x;
    int st = idx & 127;
    int jt = (idx >> 7) & 63;      // D/16
    int b  = idx >> 13;
    int s0 = st * 16, j0 = jt * 16;
    const _Float16* A = attn16 + (size_t)b * S_LEN * DMODEL;

    v8f c = {};
    for (int kc = 0; kc < DMODEL; kc += 32) {
        v16h a = load_a_frag(A, DMODEL, s0, kc, lane);
        v16h w = load_b_frag(wp16, DMODEL, j0, kc, lane);
        c = wmma_f16(a, w, c);
    }
    int n = lane & 15, hh = lane >> 4;
    float bj = bias[j0 + n];
#pragma unroll
    for (int r = 0; r < 8; ++r) {
        size_t s = (size_t)(s0 + r + 8 * hh);
        out[s * (BATCH * DMODEL) + (size_t)b * DMODEL + j0 + n] = c[r] + bj;
    }
}

// ---------------------------------------------------------------------------
extern "C" void kernel_launch(void* const* d_in, const int* in_sizes, int n_in,
                              void* d_out, int out_size, void* d_ws, size_t ws_size,
                              hipStream_t stream) {
    const float* inp   = (const float*)d_in[0];   // [S,B,D]
    const int*   amask = (const int*)d_in[1];     // [B,S]
    const float* Wqkv  = (const float*)d_in[2];   // [3D,D]
    const float* bqkv  = (const float*)d_in[3];   // [3D]
    const float* Wproj = (const float*)d_in[4];   // [D,D]
    const float* bproj = (const float*)d_in[5];   // [D]
    float* out = (float*)d_out;

    const size_t BSD = (size_t)BATCH * S_LEN * DMODEL;      // 8388608
    _Float16* ws = (_Float16*)d_ws;
    _Float16* x16     = ws;                 // BSD
    _Float16* wqkv16  = x16 + BSD;          // 3*D*D
    _Float16* wproj16 = wqkv16 + (size_t)3 * DMODEL * DMODEL;
    _Float16* q16     = wproj16 + (size_t)DMODEL * DMODEL;  // [b,h,s,dh]
    _Float16* k16     = q16 + BSD;
    _Float16* vT16    = k16 + BSD;                           // [b,h,dh,s]
    _Float16* attn16  = vT16 + BSD;                          // [b,s,d]

    // prep
    {
        int n1 = 3 * DMODEL * DMODEL;
        cvt_f16_kernel<<<(n1 + 255) / 256, 256, 0, stream>>>(Wqkv, wqkv16, n1);
        int n2 = DMODEL * DMODEL;
        cvt_f16_kernel<<<(n2 + 255) / 256, 256, 0, stream>>>(Wproj, wproj16, n2);
        int n3 = (int)BSD;
        transpose_x_kernel<<<(n3 + 255) / 256, 256, 0, stream>>>(inp, x16);
    }
    // QKV projection: (S/16) * (3D/16) * B tiles
    qkv_gemm_kernel<<<128 * 192 * BATCH, 32, 0, stream>>>(x16, wqkv16, bqkv, q16, k16, vT16);
    // Flash attention: (S/16) * H * B waves
    flash_attn_kernel<<<128 * NH * BATCH, 32, 0, stream>>>(q16, k16, vT16, amask, attn16);
    // Output projection: (S/16) * (D/16) * B tiles
    proj_gemm_kernel<<<128 * 64 * BATCH, 32, 0, stream>>>(attn16, wproj16, bproj, out);
}
